// TinyRecursiveLSTM_12799002542639
// MI455X (gfx1250) — matrix operand order repaired and checked by reading, not compiled, not run on hardware
//
#include <hip/hip_runtime.h>

typedef __attribute__((ext_vector_type(16))) _Float16 v16h;
typedef __attribute__((ext_vector_type(8)))  _Float16 v8h;
typedef __attribute__((ext_vector_type(8)))  float    v8f;
typedef __attribute__((ext_vector_type(4)))  float    v4f;

#define BATCH_   1024
#define SEQ_     512
#define IN_      64
#define HID_     64
#define NCLS_    2

#define TILE_ROWS 16
#define NWAVES    4
#define SH        72      // LDS row stride in halfs (144B, 16B aligned, bank-skewed)

// ---------- fast activations (f32, VALU transcendental path) ----------
__device__ __forceinline__ float fast_sigmoid(float x) {
    return 1.0f / (1.0f + __expf(-x));
}
__device__ __forceinline__ float fast_tanh(float x) {
    float cx = fminf(fmaxf(x, -10.0f), 10.0f);
    float e  = __expf(2.0f * cx);
    return (e - 1.0f) / (e + 1.0f);
}

// Convert 16 contiguous f32 (16B-aligned) to v16h (B-matrix lane run).
__device__ __forceinline__ v16h cvt16(const float* p) {
    const v4f* q = (const v4f*)p;
    v16h r;
#pragma unroll
    for (int c = 0; c < 4; ++c) {
        v4f t = q[c];
#pragma unroll
        for (int i = 0; i < 4; ++i) r[c * 4 + i] = (_Float16)t[i];
    }
    return r;
}

// A-operand (16x32 f16) lane fragment from a row-major f32 row in global.
// hiA = 0 for lanes 0-15, 16 for lanes 16-31.
__device__ __forceinline__ v16h loadA_g(const float* rowptr, int kc, int hiA) {
    int kb = kc * 32 + hiA;
    const v4f* p0 = (const v4f*)(rowptr + kb);        // K = kb .. kb+7
    const v4f* p1 = (const v4f*)(rowptr + kb + 8);    // K = kb+8 .. kb+15
    v16h r;
#pragma unroll
    for (int i = 0; i < 4; ++i) { r[i]     = (_Float16)p0[0][i]; r[4 + i]  = (_Float16)p0[1][i]; }
#pragma unroll
    for (int i = 0; i < 4; ++i) { r[8 + i] = (_Float16)p1[0][i]; r[12 + i] = (_Float16)p1[1][i]; }
    return r;
}

// A-operand lane fragment from a row-major f16 row in LDS (two ds_load_b128).
__device__ __forceinline__ v16h loadA_l(const _Float16* rowptr, int kc, int hiA) {
    int kb = kc * 32 + hiA;
    v8h lo = *(const v8h*)(rowptr + kb);
    v8h hi = *(const v8h*)(rowptr + kb + 8);
    v16h r;
#pragma unroll
    for (int i = 0; i < 8; ++i) { r[i] = lo[i]; r[8 + i] = hi[i]; }
    return r;
}

__global__ __launch_bounds__(128)  // 4 wave32 waves
void TinyRecursiveLSTM_kernel(const float* __restrict__ x,
                              const float* __restrict__ W_ih,
                              const float* __restrict__ W_hh,
                              const float* __restrict__ b_ih,
                              const float* __restrict__ b_hh,
                              const float* __restrict__ W_proj,
                              const float* __restrict__ b_proj,
                              const float* __restrict__ W_cls,
                              const float* __restrict__ b_cls,
                              float* __restrict__ out) {
    __shared__ _Float16 hbuf[TILE_ROWS * SH];
    __shared__ _Float16 xbuf[TILE_ROWS * SH];
    __shared__ float    h32 [TILE_ROWS * HID_];

    const int tid  = threadIdx.x;
    const int lane = tid & 31;
    const int w    = tid >> 5;                 // wave id: hidden cols [16w, 16w+16)
    const int n    = lane & 15;                // N index (col) / A row index
    const int hiA  = (lane < 16) ? 0 : 16;     // A fragment K sub-offset
    const int hiB  = (lane < 16) ? 0 : 16;     // B fragment K sub-offset
    const int mrow = (lane < 16) ? 0 : 8;      // C/D layout row offset
    const int col  = w * 16 + n;               // this lane's hidden column

    const int rowbase = blockIdx.x * TILE_ROWS;

    // ---------------- weights -> registers (f16, B-matrix layout) ----------------
    v16h b_ihw[4][2], b_hhw[4][2], b_pw[2];
#pragma unroll
    for (int g = 0; g < 4; ++g) {
        int grow = g * 64 + col;               // PyTorch gate order i,f,g,o
#pragma unroll
        for (int kc = 0; kc < 2; ++kc) {
            b_ihw[g][kc] = cvt16(W_ih + grow * IN_  + kc * 32 + hiB);
            b_hhw[g][kc] = cvt16(W_hh + grow * HID_ + kc * 32 + hiB);
        }
    }
#pragma unroll
    for (int kc = 0; kc < 2; ++kc)
        b_pw[kc] = cvt16(W_proj + col * HID_ + kc * 32 + hiB);

    float bias_g[4];
#pragma unroll
    for (int g = 0; g < 4; ++g) bias_g[g] = b_ih[g * 64 + col] + b_hh[g * 64 + col];
    const float bias_p = b_proj[col];

    // ---------------- recurrent state ----------------
    v8f c, hlast;
#pragma unroll
    for (int r = 0; r < 8; ++r) { c[r] = 0.0f; hlast[r] = 0.0f; }

    v16h a_x[2], a_h[2];
#pragma unroll
    for (int i = 0; i < 16; ++i) { a_h[0][i] = (_Float16)0.f; a_h[1][i] = (_Float16)0.f; }

    const float* xrow = x + (size_t)(rowbase + n) * SEQ_ * IN_;

    for (int t = 0; t < SEQ_; ++t) {
        // depth-0 input comes from global x, streamed + prefetched
        a_x[0] = loadA_g(xrow, 0, hiA);
        a_x[1] = loadA_g(xrow, 1, hiA);
        __builtin_prefetch(xrow + IN_, 0, 0);

#pragma unroll
        for (int d = 0; d < 3; ++d) {
            // ---- gates = x@W_ih^T + h@W_hh^T + b : 16 WMMAs ----
            v8f acc[4];
#pragma unroll
            for (int g = 0; g < 4; ++g)
#pragma unroll
                for (int r = 0; r < 8; ++r) acc[g][r] = bias_g[g];

#pragma unroll
            for (int kc = 0; kc < 2; ++kc)
#pragma unroll
                for (int g = 0; g < 4; ++g)
                    acc[g] = __builtin_amdgcn_wmma_f32_16x16x32_f16(
                        false, a_x[kc], false, b_ihw[g][kc], (short)0, acc[g], false, false);
#pragma unroll
            for (int kc = 0; kc < 2; ++kc)
#pragma unroll
                for (int g = 0; g < 4; ++g)
                    acc[g] = __builtin_amdgcn_wmma_f32_16x16x32_f16(
                        false, a_h[kc], false, b_hhw[g][kc], (short)0, acc[g], false, false);

            // ---- elementwise cell update (f32) ----
#pragma unroll
            for (int r = 0; r < 8; ++r) {
                float ig = fast_sigmoid(acc[0][r]);
                float fg = fast_sigmoid(acc[1][r]);
                float gg = fast_tanh(acc[2][r]);
                float og = fast_sigmoid(acc[3][r]);
                float cn = fg * c[r] + ig * gg;
                c[r]     = cn;
                hlast[r] = og * fast_tanh(cn);
            }

            // ---- share h across waves via LDS ----
            __syncthreads();                       // prior hbuf readers done
#pragma unroll
            for (int r = 0; r < 8; ++r)
                hbuf[(r + mrow) * SH + col] = (_Float16)hlast[r];
            __syncthreads();
            a_h[0] = loadA_l(hbuf + n * SH, 0, hiA);
            a_h[1] = loadA_l(hbuf + n * SH, 1, hiA);

            if (d < 2) {
                // ---- x_t = h @ W_proj^T + b_proj : 2 WMMAs ----
                v8f accp;
#pragma unroll
                for (int r = 0; r < 8; ++r) accp[r] = bias_p;
#pragma unroll
                for (int kc = 0; kc < 2; ++kc)
                    accp = __builtin_amdgcn_wmma_f32_16x16x32_f16(
                        false, a_h[kc], false, b_pw[kc], (short)0, accp, false, false);
#pragma unroll
                for (int r = 0; r < 8; ++r)
                    xbuf[(r + mrow) * SH + col] = (_Float16)accp[r];
                __syncthreads();
                a_x[0] = loadA_l(xbuf + n * SH, 0, hiA);
                a_x[1] = loadA_l(xbuf + n * SH, 1, hiA);
            }
        }
        xrow += IN_;
    }

    // ---------------- classifier head: logits = h @ W_cls^T + b_cls ----------------
#pragma unroll
    for (int r = 0; r < 8; ++r)
        h32[(r + mrow) * HID_ + col] = hlast[r];   // f32 copy of final h
    __syncthreads();

    if (tid < TILE_ROWS * NCLS_) {
        int r  = tid >> 1;
        int cl = tid & 1;
        float acc = b_cls[cl];
#pragma unroll
        for (int k = 0; k < HID_; ++k)
            acc += h32[r * HID_ + k] * W_cls[cl * HID_ + k];
        out[(rowbase + r) * NCLS_ + cl] = acc;
    }
}

extern "C" void kernel_launch(void* const* d_in, const int* in_sizes, int n_in,
                              void* d_out, int out_size, void* d_ws, size_t ws_size,
                              hipStream_t stream) {
    (void)in_sizes; (void)n_in; (void)out_size; (void)d_ws; (void)ws_size;
    const float* x      = (const float*)d_in[0];
    const float* W_ih   = (const float*)d_in[1];
    const float* W_hh   = (const float*)d_in[2];
    const float* b_ih   = (const float*)d_in[3];
    const float* b_hh   = (const float*)d_in[4];
    const float* W_proj = (const float*)d_in[5];
    const float* b_proj = (const float*)d_in[6];
    const float* W_cls  = (const float*)d_in[7];
    const float* b_cls  = (const float*)d_in[8];
    float* out = (float*)d_out;

    dim3 grid(BATCH_ / TILE_ROWS);   // 64 blocks
    dim3 block(128);                 // 4 wave32 waves
    TinyRecursiveLSTM_kernel<<<grid, block, 0, stream>>>(
        x, W_ih, W_hh, b_ih, b_hh, W_proj, b_proj, W_cls, b_cls, out);
}
